// ScaledDotProductAttention_59115929862674
// MI455X (gfx1250) — compile-verified, matrix operand order checked by reading, not run on hardware
//
#include <hip/hip_runtime.h>
#include <hip/hip_bf16.h>
#include <math.h>

typedef __attribute__((ext_vector_type(16))) __bf16        v16bf;
typedef __attribute__((ext_vector_type(4)))  __bf16        bf4v;
typedef __attribute__((ext_vector_type(8)))  float         v8f;
typedef __attribute__((ext_vector_type(4)))  unsigned int  uint4v;
typedef __attribute__((ext_vector_type(2)))  unsigned int  uint2v;
typedef __attribute__((ext_vector_type(4)))  float         float4v;
typedef __attribute__((ext_vector_type(4)))  int           int4v;

#define BATCH  8
#define LSEQ   2048
#define DMODEL 1024
#define SCALEF 0.03125f   /* 1/sqrt(1024) */

#define BM 128
#define BN 128
#define BK 32
#define LDSS 40           /* LDS row stride in bf16 elems: BK + 8 pad (80B, 16B aligned) */

// ---------------- bf16 conversion via packed hardware cvt -------------------
__device__ __forceinline__ uint2v cvt4_bf16(float4v f) {
    union { bf4v b; uint2v u; } p;
    p.b = __builtin_convertvector(f, bf4v);   // 2x v_cvt_pk_bf16_f32
    return p.u;
}
__device__ __forceinline__ unsigned short bf1(float x) {
    union { __bf16 h; unsigned short u; } p;
    p.h = (__bf16)x;
    return p.u;
}

// ---------------- CDNA5 async global->LDS staging ---------------------------
#if defined(__has_builtin)
#if __has_builtin(__builtin_amdgcn_global_load_async_to_lds_b128)
#define ASYNC_G2L 1
#endif
#endif
#ifndef ASYNC_G2L
#define ASYNC_G2L 0
#endif

#if ASYNC_G2L
typedef __attribute__((address_space(1))) int4v gi4;   // global int4
typedef __attribute__((address_space(3))) int4v li4;   // LDS int4
__device__ __forceinline__ void g2l_b128(const void* g, void* l) {
    __builtin_amdgcn_global_load_async_to_lds_b128((gi4*)g, (li4*)l, 0, 0);
}
#endif

__device__ __forceinline__ void async_wait0() {
#if ASYNC_G2L
#if __has_builtin(__builtin_amdgcn_s_wait_asynccnt)
    __builtin_amdgcn_s_wait_asynccnt(0);
#else
    asm volatile("s_wait_asynccnt 0x0" ::: "memory");
#endif
#endif
}

// ---------------- WMMA fragment load from LDS -------------------------------
union FragBF { v16bf bf; uint4v u[2]; };

// CDNA5 16-bit A/B fragment (wave32):
//   lanes 0-15  : row = row0+lane,    K = 0..7 and 16..23
//   lanes 16-31 : row = row0+lane-16, K = 8..15 and 24..31
__device__ __forceinline__ v16bf load_frag(const unsigned short* lds, int row0, int lane) {
    FragBF f;
    const int r  = row0 + (lane & 15);
    const int kh = (lane >> 4) << 3;          // 0 or 8
    const unsigned short* p = lds + r * LDSS;
    f.u[0] = *(const uint4v*)(p + kh);        // ds_load_b128
    f.u[1] = *(const uint4v*)(p + 16 + kh);   // ds_load_b128
    return f.bf;
}

#define WMMA_BF16(d, a, b) \
    d = __builtin_amdgcn_wmma_f32_16x16x32_bf16(false, a, false, b, (short)0, d, false, false)

// ---------------------------------------------------------------------------
// Kernel 1: Y(bf16)[L,D] = X(fp32)[L,D] @ W(fp32)[D,D]^T  (per batch, NT GEMM)
// Double-buffered: next tile prefetched to registers during WMMA.
// ---------------------------------------------------------------------------
__global__ __launch_bounds__(256) void proj_bf16_kernel(
    const float* __restrict__ X, const float* __restrict__ W,
    unsigned short* __restrict__ Y)
{
    __shared__ unsigned short sA[2][BM * LDSS];
    __shared__ unsigned short sB[2][BN * LDSS];

    const int b  = blockIdx.z;
    const int m0 = blockIdx.y * BM;
    const int n0 = blockIdx.x * BN;
    const float*    Xb = X + (size_t)b * LSEQ * DMODEL;
    unsigned short* Yb = Y + (size_t)b * LSEQ * DMODEL;

    const int t    = threadIdx.x;
    const int lane = t & 31;
    const int wave = t >> 5;
    const int wm   = wave >> 2;     // 0..1 : 64 M-rows per wave
    const int wn   = wave & 3;      // 0..3 : 32 N-cols per wave

    const v8f zero = {};
    v8f acc[4][2];
#pragma unroll
    for (int i = 0; i < 4; ++i)
#pragma unroll
        for (int j = 0; j < 2; ++j) acc[i][j] = zero;

    const int lrow = t >> 3;        // 0..31
    const int lcol = (t & 7) * 4;   // 0,4,...,28

    float4v ra[4], rb[4];

#define PROJ_LOAD(kk)                                                                   \
    _Pragma("unroll")                                                                   \
    for (int i = 0; i < 4; ++i) {                                                       \
        const int r = lrow + i * 32;                                                    \
        ra[i] = *(const float4v*)(Xb + (size_t)(m0 + r) * DMODEL + (kk) + lcol);        \
        rb[i] = *(const float4v*)(W  + (size_t)(n0 + r) * DMODEL + (kk) + lcol);        \
    }

#define PROJ_COMMIT(buf)                                                                \
    _Pragma("unroll")                                                                   \
    for (int i = 0; i < 4; ++i) {                                                       \
        const int r = lrow + i * 32;                                                    \
        *(uint2v*)(sA[buf] + r * LDSS + lcol) = cvt4_bf16(ra[i]);                       \
        *(uint2v*)(sB[buf] + r * LDSS + lcol) = cvt4_bf16(rb[i]);                       \
    }

    PROJ_LOAD(0);
    PROJ_COMMIT(0);
    __syncthreads();

    for (int kk = 0; kk < DMODEL; kk += BK) {
        const int  cur       = (kk >> 5) & 1;
        const bool have_next = (kk + BK) < DMODEL;
        if (have_next) PROJ_LOAD(kk + BK);

        v16bf bfr[2];
#pragma unroll
        for (int j = 0; j < 2; ++j) bfr[j] = load_frag(sB[cur], wn * 32 + j * 16, lane);
#pragma unroll
        for (int i = 0; i < 4; ++i) {
            v16bf afr = load_frag(sA[cur], wm * 64 + i * 16, lane);
#pragma unroll
            for (int j = 0; j < 2; ++j) WMMA_BF16(acc[i][j], afr, bfr[j]);
        }

        if (have_next) PROJ_COMMIT(cur ^ 1);
        __syncthreads();
    }

    // C layout: VGPR v -> M = v + 8*(lane>=16), N = lane&15
    const int cn = lane & 15;
    const int rb2 = (lane >> 4) * 8;
#pragma unroll
    for (int i = 0; i < 4; ++i)
#pragma unroll
        for (int j = 0; j < 2; ++j) {
            const int n = n0 + wn * 32 + j * 16 + cn;
#pragma unroll
            for (int v = 0; v < 8; ++v) {
                const int m = m0 + wm * 64 + i * 16 + rb2 + v;
                Yb[(size_t)m * DMODEL + n] = bf1(acc[i][j][v]);
            }
        }
#undef PROJ_LOAD
#undef PROJ_COMMIT
}

// ---------------------------------------------------------------------------
// Kernel 2: S(fp32)[L,L] = (Q(bf16)[L,D] @ K(bf16)[L,D]^T) * SCALE, masked
// Double-buffered with async global->LDS staging (ASYNCcnt).
// ---------------------------------------------------------------------------
__global__ __launch_bounds__(256) void scores_kernel(
    const unsigned short* __restrict__ Q, const unsigned short* __restrict__ K,
    const int* __restrict__ mask, float* __restrict__ S)
{
    __shared__ unsigned short sA[2][BM * LDSS];
    __shared__ unsigned short sB[2][BN * LDSS];

    const int b  = blockIdx.z;
    const int m0 = blockIdx.y * BM;
    const int n0 = blockIdx.x * BN;
    const unsigned short* Qb = Q + (size_t)b * LSEQ * DMODEL;
    const unsigned short* Kb = K + (size_t)b * LSEQ * DMODEL;
    const int*   maskb = mask + (size_t)b * LSEQ;
    float*       Sb    = S + (size_t)b * LSEQ * LSEQ;

    const int t    = threadIdx.x;
    const int lane = t & 31;
    const int wave = t >> 5;
    const int wm   = wave >> 2;
    const int wn   = wave & 3;

    const v8f zero = {};
    v8f acc[4][2];
#pragma unroll
    for (int i = 0; i < 4; ++i)
#pragma unroll
        for (int j = 0; j < 2; ++j) acc[i][j] = zero;

    const int lrow = t >> 2;        // 0..63
    const int lcol = (t & 3) * 8;   // 0,8,16,24

#if ASYNC_G2L
#define SC_COPY(g, l) g2l_b128(g, l)
#else
#define SC_COPY(g, l) (*(uint4v*)(l) = *(const uint4v*)(g))
#endif

#define SC_STAGE(buf, kk)                                                               \
    _Pragma("unroll")                                                                   \
    for (int i = 0; i < 2; ++i) {                                                       \
        const int r = lrow + i * 64;                                                    \
        const void* ga = Qb + (size_t)(m0 + r) * DMODEL + (kk) + lcol;                  \
        const void* gb = Kb + (size_t)(n0 + r) * DMODEL + (kk) + lcol;                  \
        void* la = sA[buf] + r * LDSS + lcol;                                           \
        void* lb = sB[buf] + r * LDSS + lcol;                                           \
        SC_COPY(ga, la); SC_COPY(gb, lb);                                               \
    }

    SC_STAGE(0, 0);
    async_wait0();
    __syncthreads();

    for (int kk = 0; kk < DMODEL; kk += BK) {
        const int  cur       = (kk >> 5) & 1;
        const bool have_next = (kk + BK) < DMODEL;
        if (have_next) SC_STAGE(cur ^ 1, kk + BK);   // fire-and-forget async copies

        v16bf bfr[2];
#pragma unroll
        for (int j = 0; j < 2; ++j) bfr[j] = load_frag(sB[cur], wn * 32 + j * 16, lane);
#pragma unroll
        for (int i = 0; i < 4; ++i) {
            v16bf afr = load_frag(sA[cur], wm * 64 + i * 16, lane);
#pragma unroll
            for (int j = 0; j < 2; ++j) WMMA_BF16(acc[i][j], afr, bfr[j]);
        }

        async_wait0();                               // drain copies into nxt buffer
        __syncthreads();
    }

    const int cn  = lane & 15;
    const int rb2 = (lane >> 4) * 8;
#pragma unroll
    for (int i = 0; i < 4; ++i)
#pragma unroll
        for (int j = 0; j < 2; ++j) {
            const int n = n0 + wn * 32 + j * 16 + cn;
            const bool keep = (maskb[n] != 0);
#pragma unroll
            for (int v = 0; v < 8; ++v) {
                const int m = m0 + wm * 64 + i * 16 + rb2 + v;
                Sb[(size_t)m * LSEQ + n] = keep ? acc[i][j][v] * SCALEF : -INFINITY;
            }
        }
#undef SC_STAGE
#undef SC_COPY
}

// ---------------------------------------------------------------------------
// Kernel 3: in-place row softmax over LSEQ (one 256-thread block per row)
// ---------------------------------------------------------------------------
__global__ __launch_bounds__(256) void softmax_kernel(float* __restrict__ S)
{
    __shared__ float red[256];
    float* p = S + (size_t)blockIdx.x * LSEQ;
    const int t = threadIdx.x;

    float x[8];
    float m = -INFINITY;
#pragma unroll
    for (int i = 0; i < 8; ++i) { x[i] = p[t + i * 256]; m = fmaxf(m, x[i]); }
    red[t] = m; __syncthreads();
    for (int s = 128; s > 0; s >>= 1) {
        if (t < s) red[t] = fmaxf(red[t], red[t + s]);
        __syncthreads();
    }
    m = red[0]; __syncthreads();

    float sum = 0.f;
#pragma unroll
    for (int i = 0; i < 8; ++i) { x[i] = __expf(x[i] - m); sum += x[i]; }
    red[t] = sum; __syncthreads();
    for (int s = 128; s > 0; s >>= 1) {
        if (t < s) red[t] = red[t] + red[t + s];
        __syncthreads();
    }
    const float inv = 1.0f / red[0];
#pragma unroll
    for (int i = 0; i < 8; ++i) p[t + i * 256] = x[i] * inv;
}

// ---------------------------------------------------------------------------
// Kernel 4: O(fp32)[L,D] = P(fp32)[L,L] @ V(fp32)[L,D]  (V transposed in LDS)
// Double-buffered with register prefetch.
// ---------------------------------------------------------------------------
__global__ __launch_bounds__(256) void context_kernel(
    const float* __restrict__ P, const float* __restrict__ V,
    float* __restrict__ O)
{
    __shared__ unsigned short sA[2][BM * LDSS];
    __shared__ unsigned short sB[2][BN * LDSS];

    const int b  = blockIdx.z;
    const int m0 = blockIdx.y * BM;
    const int n0 = blockIdx.x * BN;
    const float* Pb = P + (size_t)b * LSEQ * LSEQ;
    const float* Vb = V + (size_t)b * LSEQ * DMODEL;
    float*       Ob = O + (size_t)b * LSEQ * DMODEL;

    const int t    = threadIdx.x;
    const int lane = t & 31;
    const int wave = t >> 5;
    const int wm   = wave >> 2;
    const int wn   = wave & 3;

    const v8f zero = {};
    v8f acc[4][2];
#pragma unroll
    for (int i = 0; i < 4; ++i)
#pragma unroll
        for (int j = 0; j < 2; ++j) acc[i][j] = zero;

    const int lrow = t >> 3;        // 0..31
    const int lcol = (t & 7) * 4;   // 0..28
    const int kO   = t >> 3;        // 0..31 (B-tile k row)
    const int nb   = (t & 7) * 4;   // 0..28 (B-tile n base)

    float4v ra[4], rv[4];

#define CTX_LOAD(kk)                                                                    \
    _Pragma("unroll")                                                                   \
    for (int i = 0; i < 4; ++i) {                                                       \
        ra[i] = *(const float4v*)(Pb + (size_t)(m0 + lrow + i * 32) * LSEQ + (kk) + lcol); \
        rv[i] = *(const float4v*)(Vb + (size_t)((kk) + kO) * DMODEL + n0 + nb + i * 32);   \
    }

#define CTX_COMMIT(buf)                                                                 \
    _Pragma("unroll")                                                                   \
    for (int i = 0; i < 4; ++i) {                                                       \
        const int r = lrow + i * 32;                                                    \
        *(uint2v*)(sA[buf] + r * LDSS + lcol) = cvt4_bf16(ra[i]);                       \
        const int n = nb + i * 32;                                                      \
        sB[buf][(n + 0) * LDSS + kO] = bf1(rv[i].x);                                    \
        sB[buf][(n + 1) * LDSS + kO] = bf1(rv[i].y);                                    \
        sB[buf][(n + 2) * LDSS + kO] = bf1(rv[i].z);                                    \
        sB[buf][(n + 3) * LDSS + kO] = bf1(rv[i].w);                                    \
    }

    CTX_LOAD(0);
    CTX_COMMIT(0);
    __syncthreads();

    for (int kk = 0; kk < LSEQ; kk += BK) {
        const int  cur       = (kk >> 5) & 1;
        const bool have_next = (kk + BK) < LSEQ;
        if (have_next) CTX_LOAD(kk + BK);

        v16bf bfr[2];
#pragma unroll
        for (int j = 0; j < 2; ++j) bfr[j] = load_frag(sB[cur], wn * 32 + j * 16, lane);
#pragma unroll
        for (int i = 0; i < 4; ++i) {
            v16bf afr = load_frag(sA[cur], wm * 64 + i * 16, lane);
#pragma unroll
            for (int j = 0; j < 2; ++j) WMMA_BF16(acc[i][j], afr, bfr[j]);
        }

        if (have_next) CTX_COMMIT(cur ^ 1);
        __syncthreads();
    }

    const int cn  = lane & 15;
    const int rb2 = (lane >> 4) * 8;
#pragma unroll
    for (int i = 0; i < 4; ++i)
#pragma unroll
        for (int j = 0; j < 2; ++j) {
            const int n = n0 + wn * 32 + j * 16 + cn;
#pragma unroll
            for (int v = 0; v < 8; ++v) {
                const int m = m0 + wm * 64 + i * 16 + rb2 + v;
                Ob[(size_t)m * DMODEL + n] = acc[i][j][v];
            }
        }
#undef CTX_LOAD
#undef CTX_COMMIT
}

// ---------------------------------------------------------------------------
extern "C" void kernel_launch(void* const* d_in, const int* in_sizes, int n_in,
                              void* d_out, int out_size, void* d_ws, size_t ws_size,
                              hipStream_t stream) {
    const float* query = (const float*)d_in[0];
    const float* key   = (const float*)d_in[1];
    const float* value = (const float*)d_in[2];
    const int*   mask  = (const int*)d_in[3];
    const float* Wq    = (const float*)d_in[4];
    const float* Wk    = (const float*)d_in[5];

    float* attn_cxt = (float*)d_out;                                   // [B,LQ,DV]
    float* attn_w   = (float*)d_out + (size_t)BATCH * LSEQ * DMODEL;   // [B,LQ,LK]

    unsigned short* qb = (unsigned short*)d_ws;                        // bf16 q
    unsigned short* kb = qb + (size_t)BATCH * LSEQ * DMODEL;           // bf16 k

    dim3 blk(256);
    dim3 gproj(DMODEL / BN, LSEQ / BM, BATCH);   // (8,16,8)
    dim3 gsc  (LSEQ / BN,  LSEQ / BM, BATCH);    // (16,16,8)
    dim3 gctx (DMODEL / BN, LSEQ / BM, BATCH);   // (8,16,8)

    proj_bf16_kernel<<<gproj, blk, 0, stream>>>(query, Wq, qb);
    proj_bf16_kernel<<<gproj, blk, 0, stream>>>(key,   Wk, kb);
    scores_kernel  <<<gsc,   blk, 0, stream>>>(qb, kb, mask, attn_w);
    softmax_kernel <<<dim3(BATCH * LSEQ), blk, 0, stream>>>(attn_w);
    context_kernel <<<gctx,  blk, 0, stream>>>(attn_w, value, attn_cxt);
}
